// AGATBD_72662256714421
// MI455X (gfx1250) — matrix operand order repaired and checked by reading, not compiled, not run on hardware
//
#include <hip/hip_runtime.h>
#include <cstddef>
#include <cstdint>

// ---------------------------------------------------------------------------
// Types / WMMA helpers (CDNA5 / gfx1250, wave32)
// ---------------------------------------------------------------------------
typedef __attribute__((ext_vector_type(16))) _Float16 v16h;
typedef __attribute__((ext_vector_type(8)))  float    v8f;
typedef __attribute__((__vector_size__(4 * sizeof(int)))) int vi4;

#define DEVINL __device__ __forceinline__

// Async global->LDS staging (CDNA5): gated on builtin availability so the
// kernel still compiles (with manual staging) if this toolchain lacks it.
#if defined(__has_builtin)
#  if __has_builtin(__builtin_amdgcn_global_load_async_to_lds_b128)
#    define ASYNC_LDS 1
#  endif
#endif
#ifndef ASYNC_LDS
#  define ASYNC_LDS 0
#endif

#if ASYNC_LDS
DEVINL void async_copy16(const _Float16* g, _Float16* l) {
  __builtin_amdgcn_global_load_async_to_lds_b128(
      (__attribute__((address_space(1))) vi4*)g,
      (__attribute__((address_space(3))) vi4*)l, 0, 0);
}
DEVINL void async_wait0() {
#  if __has_builtin(__builtin_amdgcn_s_wait_asynccnt)
  __builtin_amdgcn_s_wait_asynccnt(0);
#  else
  asm volatile("s_wait_asynccnt 0" ::: "memory");
#  endif
}
#endif

DEVINL v8f wmma_f16(v16h a, v16h b, v8f c) {
  // D = A(16x32 f16) x B(32x16 f16) + C(16x16 f32)
  return __builtin_amdgcn_wmma_f32_16x16x32_f16(false, a, false, b, (short)0, c,
                                                false, false);
}

// A fragment: 16x32 f16, row-major source with row stride `stride` (halfs).
// Layout: lanes 0-15 -> K 0..15 ; lanes 16-31 -> K 16..31 (contiguous, matches frag_b).
DEVINL v16h frag_a(const _Float16* base, int stride, int lane) {
  const _Float16* p = base + (size_t)(lane & 15) * stride + ((lane >> 4) << 4);
  v16h a;
#pragma unroll
  for (int e = 0; e < 16; ++e) a[e] = p[e];
  return a;
}

// B fragment from transposed weights Bt (N,K): lane holds column n = lane&15,
// 16 contiguous K values starting at (lane>>4)*16.
DEVINL v16h frag_b(const _Float16* base, int stride, int lane) {
  const _Float16* p = base + (size_t)(lane & 15) * stride + ((lane >> 4) << 4);
  v16h b;
#pragma unroll
  for (int e = 0; e < 16; ++e) b[e] = p[e];
  return b;
}

DEVINL v16h frag_b_guard(const _Float16* base, int stride, int nvalid, int lane) {
  int n = lane & 15;
  v16h b;
  if (n < nvalid) {
    const _Float16* p = base + (size_t)n * stride + ((lane >> 4) << 4);
#pragma unroll
    for (int e = 0; e < 16; ++e) b[e] = p[e];
  } else {
#pragma unroll
    for (int e = 0; e < 16; ++e) b[e] = (_Float16)0.f;
  }
  return b;
}

// ---------------------------------------------------------------------------
// Model constants
// ---------------------------------------------------------------------------
static constexpr int BB   = 64;      // batch
static constexpr int NN   = 512;     // nodes
static constexpr int SEQ  = 256;
static constexpr int DLEN = 4096;
static constexpr int EMB  = 100;
static constexpr int HFE  = 128;     // gat1 per-head feats
static constexpr int OFE  = 64;      // gat2 per-head feats (= GRU timesteps)
static constexpr int HID  = 256;
static constexpr int OUTD = 12;
static constexpr int L1   = DLEN - 9;          // 4087
static constexpr int L2   = L1 - 9;            // 4078
static constexpr int DIMFC = 16 * L2;          // 65248 (mult of 32)
static constexpr int MLP_IN  = OFE + HID;      // 320
static constexpr int MLP_MID = MLP_IN / 2;     // 160

// ---------------------------------------------------------------------------
// Generic tiled WMMA GEMM:  C(MxN) = act(A(MxK,f16) * Bt(NxK,f16)^T + bias)
// Block tile 64x64, 8 waves, 2 16x16 tiles/wave, K step 32.
// Tile staging uses async global->LDS copies when the toolchain exposes them.
// ---------------------------------------------------------------------------
static __global__ __launch_bounds__(256)
void k_gemm(const _Float16* __restrict__ A, const _Float16* __restrict__ Bt,
            float* __restrict__ Cf, _Float16* __restrict__ Ch,
            const float* __restrict__ bias, int M, int N, int K, int act)
{
  __shared__ _Float16 As[64 * 40];
  __shared__ _Float16 Bs[64 * 40];
  const int tid = threadIdx.x, lane = tid & 31, wave = tid >> 5;
  const int m0 = blockIdx.y * 64, n0 = blockIdx.x * 64;
  const int mt = wave >> 1, ng = (wave & 1) * 2;
  v8f acc0 = {}, acc1 = {};
  const int lrow = tid >> 2, lcol = (tid & 3) * 8;
  const int gm = m0 + lrow, gn = n0 + lrow;

  for (int kk = 0; kk < K; kk += 32) {
    _Float16* da = As + lrow * 40 + lcol;
    _Float16* db = Bs + lrow * 40 + lcol;
#if ASYNC_LDS
    if (gm < M) {
      async_copy16(A + (size_t)gm * K + kk + lcol, da);
    } else {
#pragma unroll
      for (int e = 0; e < 8; ++e) da[e] = (_Float16)0.f;
    }
    if (gn < N) {
      async_copy16(Bt + (size_t)gn * K + kk + lcol, db);
    } else {
#pragma unroll
      for (int e = 0; e < 8; ++e) db[e] = (_Float16)0.f;
    }
    async_wait0();
#else
    if (gm < M) {
      const _Float16* s = A + (size_t)gm * K + kk + lcol;
      __builtin_prefetch(s + 256, 0, 1);
#pragma unroll
      for (int e = 0; e < 8; ++e) da[e] = s[e];
    } else {
#pragma unroll
      for (int e = 0; e < 8; ++e) da[e] = (_Float16)0.f;
    }
    if (gn < N) {
      const _Float16* s = Bt + (size_t)gn * K + kk + lcol;
      __builtin_prefetch(s + 256, 0, 1);
#pragma unroll
      for (int e = 0; e < 8; ++e) db[e] = s[e];
    } else {
#pragma unroll
      for (int e = 0; e < 8; ++e) db[e] = (_Float16)0.f;
    }
#endif
    __syncthreads();
    v16h af = frag_a(As + mt * 16 * 40, 40, lane);
    v16h b0 = frag_b(Bs + (ng + 0) * 16 * 40, 40, lane);
    v16h b1 = frag_b(Bs + (ng + 1) * 16 * 40, 40, lane);
    acc0 = wmma_f16(af, b0, acc0);
    acc1 = wmma_f16(af, b1, acc1);
    __syncthreads();
  }
  const int rbase = m0 + mt * 16 + ((lane >> 4) << 3);
#pragma unroll
  for (int h = 0; h < 2; ++h) {
    v8f acc = h ? acc1 : acc0;
    int col = n0 + (ng + h) * 16 + (lane & 15);
    if (col < N) {
      float bv = bias ? bias[col] : 0.f;
#pragma unroll
      for (int v = 0; v < 8; ++v) {
        int row = rbase + v;
        if (row < M) {
          float x = acc[v] + bv;
          if (act) x = fmaxf(x, 0.f);
          if (Cf) Cf[(size_t)row * N + col] = x;
          if (Ch) Ch[(size_t)row * N + col] = (_Float16)x;
        }
      }
    }
  }
}

// ---------------------------------------------------------------------------
// Small utility kernels
// ---------------------------------------------------------------------------
static __global__ void k_zero(float* p, int n) {
  int i = blockIdx.x * blockDim.x + threadIdx.x;
  if (i < n) p[i] = 0.f;
}
static __global__ void k_copy(const float* __restrict__ s, float* __restrict__ d, int n) {
  int i = blockIdx.x * blockDim.x + threadIdx.x;
  if (i < n) d[i] = s[i];
}
static __global__ void k_convert_h(const float* __restrict__ s, _Float16* __restrict__ d, int n) {
  int i = blockIdx.x * blockDim.x + threadIdx.x;
  if (i < n) d[i] = (_Float16)s[i];
}
// src (Nb,R,C) f32 -> dst (Nb,C,R) f16   (builds Bt weight layouts)
static __global__ void k_transpose_b(const float* __restrict__ src, _Float16* __restrict__ dst,
                                     int Nb, int R, int C) {
  int i = blockIdx.x * blockDim.x + threadIdx.x;
  int total = Nb * R * C;
  if (i >= total) return;
  int r  = i % R;
  int c  = (i / R) % C;
  int nb = i / (R * C);
  dst[i] = (_Float16)src[(size_t)nb * R * C + (size_t)r * C + c];
}

// ---------------------------------------------------------------------------
// Conv stack + batch norms
// ---------------------------------------------------------------------------
static __global__ __launch_bounds__(256)
void k_conv1(const float* __restrict__ x, const float* __restrict__ w,
             const float* __restrict__ b, float* __restrict__ z1)
{
  __shared__ float sw[80], sb[8];
  if (threadIdx.x < 80) sw[threadIdx.x] = w[threadIdx.x];
  if (threadIdx.x < 8)  sb[threadIdx.x] = b[threadIdx.x];
  __syncthreads();
  int idx = blockIdx.x * 256 + threadIdx.x;
  if (idx >= NN * L1) return;
  int n = idx / L1, l = idx % L1;
  float xv[10];
#pragma unroll
  for (int t = 0; t < 10; ++t) xv[t] = x[(size_t)n * DLEN + l + t];
#pragma unroll
  for (int c = 0; c < 8; ++c) {
    float a = sb[c];
#pragma unroll
    for (int t = 0; t < 10; ++t) a += sw[c * 10 + t] * xv[t];
    z1[((size_t)n * 8 + c) * L1 + l] = fmaxf(a, 0.f);
  }
}

// per-(row,channel) partial sums -> atomic accumulate per channel
static __global__ __launch_bounds__(256)
void k_bn_stats(const float* __restrict__ x, float* __restrict__ sums,
                float* __restrict__ sqs, int C, int L)
{
  __shared__ float r1[256], r2[256];
  int blk = blockIdx.x, tid = threadIdx.x;
  int c = blk % C;
  const float* p = x + (size_t)blk * L;
  float s = 0.f, q = 0.f;
  for (int i = tid; i < L; i += 256) { float v = p[i]; s += v; q += v * v; }
  r1[tid] = s; r2[tid] = q; __syncthreads();
  for (int st = 128; st > 0; st >>= 1) {
    if (tid < st) { r1[tid] += r1[tid + st]; r2[tid] += r2[tid + st]; }
    __syncthreads();
  }
  if (tid == 0) { atomicAdd(&sums[c], r1[0]); atomicAdd(&sqs[c], r2[0]); }
}

// stats over dim0 for (R=512, C=100) matrix
static __global__ __launch_bounds__(256)
void k_bn3_stats(const float* __restrict__ z, float* __restrict__ sums, float* __restrict__ sqs)
{
  __shared__ float r1[256], r2[256];
  int c = blockIdx.x, tid = threadIdx.x;
  float s = 0.f, q = 0.f;
  for (int n = tid; n < NN; n += 256) { float v = z[(size_t)n * EMB + c]; s += v; q += v * v; }
  r1[tid] = s; r2[tid] = q; __syncthreads();
  for (int st = 128; st > 0; st >>= 1) {
    if (tid < st) { r1[tid] += r1[tid + st]; r2[tid] += r2[tid + st]; }
    __syncthreads();
  }
  if (tid == 0) { sums[c] = r1[0]; sqs[c] = r2[0]; }
}

// region layout (floats): [0]=sums [128]=sqs [256]=scale [384]=shift
static __global__ void k_bn_finalize(float* __restrict__ reg, const float* __restrict__ g,
                                     const float* __restrict__ b, float cnt, int C)
{
  int c = threadIdx.x;
  if (c < C) {
    float m   = reg[c] / cnt;
    float var = reg[128 + c] / cnt - m * m;
    float sc  = g[c] * rsqrtf(var + 1e-5f);
    reg[256 + c] = sc;
    reg[384 + c] = b[c] - m * sc;
  }
}

static __global__ __launch_bounds__(256)
void k_conv2(const float* __restrict__ z1, const float* __restrict__ bn1,
             const float* __restrict__ w, const float* __restrict__ b,
             float* __restrict__ z2)
{
  __shared__ float sw[1280], sb[16], ssc[8], ssh[8];
  for (int t = threadIdx.x; t < 1280; t += 256) sw[t] = w[t];
  if (threadIdx.x < 16) sb[threadIdx.x] = b[threadIdx.x];
  if (threadIdx.x < 8)  { ssc[threadIdx.x] = bn1[256 + threadIdx.x]; ssh[threadIdx.x] = bn1[384 + threadIdx.x]; }
  __syncthreads();
  int idx = blockIdx.x * 256 + threadIdx.x;
  if (idx >= NN * L2) return;
  int n = idx / L2, l = idx % L2;
  float acc[16];
#pragma unroll
  for (int c = 0; c < 16; ++c) acc[c] = sb[c];
  for (int ci = 0; ci < 8; ++ci) {
    float scv = ssc[ci], shv = ssh[ci];
    float v[10];
#pragma unroll
    for (int t = 0; t < 10; ++t)
      v[t] = z1[((size_t)n * 8 + ci) * L1 + l + t] * scv + shv;
#pragma unroll
    for (int c = 0; c < 16; ++c) {
      float a = 0.f;
#pragma unroll
      for (int t = 0; t < 10; ++t) a += sw[(c * 8 + ci) * 10 + t] * v[t];
      acc[c] += a;
    }
  }
#pragma unroll
  for (int c = 0; c < 16; ++c)
    z2[((size_t)n * 16 + c) * L2 + l] = fmaxf(acc[c], 0.f);
}

// normalized z2 -> f16 GEMM A operand (flat (N, 16*L2) row-major)
static __global__ void k_z2h(const float* __restrict__ z2, const float* __restrict__ bn2,
                             _Float16* __restrict__ z2h, int total)
{
  int i = blockIdx.x * blockDim.x + threadIdx.x;
  if (i >= total) return;
  int c = (i / L2) % 16;
  z2h[i] = (_Float16)(z2[i] * bn2[256 + c] + bn2[384 + c]);
}

// bn3-normalize z (512x100) into f16 padded (512x128)
static __global__ void k_z16pad(const float* __restrict__ z, const float* __restrict__ bn3,
                                _Float16* __restrict__ z16)
{
  int i = blockIdx.x * blockDim.x + threadIdx.x;
  if (i >= NN * 128) return;
  int n = i >> 7, k = i & 127;
  z16[i] = (k < EMB) ? (_Float16)(z[(size_t)n * EMB + k] * bn3[256 + k] + bn3[384 + k])
                     : (_Float16)0.f;
}

// split fco_w (200x100) into transposed, K-padded halves UtW/VtW (100x128 f16)
static __global__ void k_uvw(const float* __restrict__ fco_w,
                             _Float16* __restrict__ UtW, _Float16* __restrict__ VtW)
{
  int i = blockIdx.x * blockDim.x + threadIdx.x;
  if (i >= EMB * 128) return;
  int o = i >> 7, k = i & 127;
  UtW[i] = (k < EMB) ? (_Float16)fco_w[(size_t)k * EMB + o] : (_Float16)0.f;
  VtW[i] = (k < EMB) ? (_Float16)fco_w[(size_t)(EMB + k) * EMB + o] : (_Float16)0.f;
}

// ---------------------------------------------------------------------------
// Pair MLP + gumbel straight-through -> adjacency (binary, diag forced to 1)
// hp[i,j] = relu(U[j] + V[i] + fco_b);  logits = hp @ fcc_w + fcc_b
// ---------------------------------------------------------------------------
static __global__ __launch_bounds__(256)
void k_pair(const float* __restrict__ U, const float* __restrict__ V,
            const float* __restrict__ fco_b, const float* __restrict__ fcc_w,
            const float* __restrict__ fcc_b, const float* __restrict__ gu,
            float* __restrict__ adj)
{
  __shared__ float sU[16 * EMB], sV[16 * EMB], sb[EMB], sw0[EMB], sw1[EMB];
  int i0 = blockIdx.y * 16, j0 = blockIdx.x * 16, tid = threadIdx.x;
  for (int t = tid; t < 16 * EMB; t += 256) {
    int r = t / EMB, k = t % EMB;
    sU[t] = U[(size_t)(j0 + r) * EMB + k];
    sV[t] = V[(size_t)(i0 + r) * EMB + k];
  }
  for (int t = tid; t < EMB; t += 256) {
    sb[t] = fco_b[t]; sw0[t] = fcc_w[t * 2]; sw1[t] = fcc_w[t * 2 + 1];
  }
  __syncthreads();
  int li = tid >> 4, lj = tid & 15;
  int i = i0 + li, j = j0 + lj;
  float l0 = fcc_b[0], l1 = fcc_b[1];
  for (int k = 0; k < EMB; ++k) {
    float hp = fmaxf(sV[li * EMB + k] + sU[lj * EMB + k] + sb[k], 0.f);
    l0 += hp * sw0[k]; l1 += hp * sw1[k];
  }
  float u0 = gu[((size_t)i * NN + j) * 2 + 0];
  float u1 = gu[((size_t)i * NN + j) * 2 + 1];
  float g0 = -__logf(-__logf(u0 + 1e-20f) + 1e-20f);
  float g1 = -__logf(-__logf(u1 + 1e-20f) + 1e-20f);
  adj[(size_t)i * NN + j] = (i == j) ? 1.f : ((l0 + g0 >= l1 + g1) ? 1.f : 0.f);
}

// ---------------------------------------------------------------------------
// GAT pieces
// ---------------------------------------------------------------------------
// asrc/adst coefficients: one thread per (b,n,head)
static __global__ void k_attn_coefs(const _Float16* __restrict__ h16,
                                    const float* __restrict__ avs, const float* __restrict__ avd,
                                    float* __restrict__ asrc, float* __restrict__ adst, int D)
{
  int idx = blockIdx.x * blockDim.x + threadIdx.x;
  if (idx >= BB * NN * 4) return;
  int hh = idx & 3;
  const _Float16* hp = h16 + (size_t)idx * D;
  float s1 = 0.f, s2 = 0.f;
  for (int d = 0; d < D; ++d) {
    float hv = (float)hp[d];
    s1 += hv * avs[hh * D + d];
    s2 += hv * avd[hh * D + d];
  }
  asrc[idx] = s1; adst[idx] = s2;
}

// softmax stats over source axis i for each (b, j, head)
static __global__ __launch_bounds__(128)
void k_gat_stats(const float* __restrict__ asrc, const float* __restrict__ adst,
                 const float* __restrict__ adj, float* __restrict__ mrow,
                 float* __restrict__ srow)
{
  __shared__ float red[128];
  int j = blockIdx.x, b = blockIdx.y, tid = threadIdx.x;
  for (int hh = 0; hh < 4; ++hh) {
    float aj = adst[((size_t)b * NN + j) * 4 + hh];
    float mx = -1e30f;
    for (int i = tid; i < NN; i += 128) {
      if (adj[(size_t)i * NN + j] > 0.f) {
        float e  = asrc[((size_t)b * NN + i) * 4 + hh] + aj;
        float lr = e > 0.f ? e : 0.2f * e;
        mx = fmaxf(mx, lr);
      }
    }
    red[tid] = mx; __syncthreads();
    for (int st = 64; st > 0; st >>= 1) {
      if (tid < st) red[tid] = fmaxf(red[tid], red[tid + st]);
      __syncthreads();
    }
    mx = red[0]; __syncthreads();
    float sm = 0.f;
    for (int i = tid; i < NN; i += 128) {
      if (adj[(size_t)i * NN + j] > 0.f) {
        float e  = asrc[((size_t)b * NN + i) * 4 + hh] + aj;
        float lr = e > 0.f ? e : 0.2f * e;
        sm += __expf(lr - mx);
      }
    }
    red[tid] = sm; __syncthreads();
    for (int st = 64; st > 0; st >>= 1) {
      if (tid < st) red[tid] += red[tid + st];
      __syncthreads();
    }
    if (tid == 0) {
      mrow[((size_t)b * NN + j) * 4 + hh] = mx;
      srow[((size_t)b * NN + j) * 4 + hh] = red[0];
    }
    __syncthreads();
  }
}

// Flash-style masked-softmax aggregation with WMMA:
// out[b,j,h,:] = sum_i exp(lrelu(asrc_i+adst_j)-m_j)*mask(i,j) * hval[b,i,h,:] / s_j
template <int D>
static __global__ __launch_bounds__(256)
void k_gat_attn(const _Float16* __restrict__ h16, const float* __restrict__ asrc,
                const float* __restrict__ adst, const float* __restrict__ mrow,
                const float* __restrict__ srow, const float* __restrict__ adj,
                float* __restrict__ att)
{
  constexpr int TPD = D / 32;                 // d-tiles per wave
  __shared__ _Float16 Aw[64 * 40];            // weights (j x i) tile
  __shared__ _Float16 Bsd[D * 40];            // values transposed (d x i)
  __shared__ float sAdst[64], sM[64];
  const int b = blockIdx.z, hh = blockIdx.y, j0 = blockIdx.x * 64;
  const int tid = threadIdx.x, lane = tid & 31, wave = tid >> 5;
  if (tid < 64) {
    int gj = j0 + tid;
    sAdst[tid] = adst[((size_t)b * NN + gj) * 4 + hh];
    sM[tid]    = mrow[((size_t)b * NN + gj) * 4 + hh];
  }
  v8f acc[TPD] = {};
  const int jt  = wave >> 1;
  const int dt0 = (wave & 1) * TPD;

  for (int ii = 0; ii < NN; ii += 32) {
    __syncthreads();
    // attention-weight tile Aw[j][i]
    {
      int j  = tid >> 2;
      int ib = (tid & 3) * 8;
      float aj = sAdst[j], mj = sM[j];
#pragma unroll
      for (int e = 0; e < 8; ++e) {
        int i = ib + e;
        float am = adj[(size_t)(ii + i) * NN + (j0 + j)];
        float ev = asrc[((size_t)b * NN + ii + i) * 4 + hh] + aj;
        float lr = ev > 0.f ? ev : 0.2f * ev;
        float w  = (am > 0.f) ? __expf(lr - mj) : 0.f;
        Aw[j * 40 + i] = (_Float16)w;
      }
    }
    // value chunk transposed into Bsd[d][i]
    {
      int i  = tid >> 3;
      int db = (tid & 7) * (D / 8);
      const _Float16* src = h16 + ((size_t)(b * NN + ii + i) * 4 + hh) * D + db;
#pragma unroll
      for (int e = 0; e < D / 8; ++e) Bsd[(db + e) * 40 + i] = src[e];
    }
    __syncthreads();
    v16h af = frag_a(Aw + jt * 16 * 40, 40, lane);
#pragma unroll
    for (int q = 0; q < TPD; ++q) {
      v16h bf = frag_b(Bsd + (dt0 + q) * 16 * 40, 40, lane);
      acc[q] = wmma_f16(af, bf, acc[q]);
    }
  }
  const int jrow = j0 + jt * 16 + ((lane >> 4) << 3);
  const int dc0  = dt0 * 16 + (lane & 15);
#pragma unroll
  for (int q = 0; q < TPD; ++q) {
    int d = dc0 + q * 16;
#pragma unroll
    for (int v = 0; v < 8; ++v) {
      int j = jrow + v;
      float sv = srow[((size_t)b * NN + j) * 4 + hh];
      att[((size_t)(b * NN + j) * 4 + hh) * D + d] = acc[q][v] / sv;
    }
  }
}

// head-mean + bias (+ optional ELU); writes f32 and optional f16 copy
static __global__ void k_combine(const float* __restrict__ att, const float* __restrict__ bias,
                                 float* __restrict__ outf, _Float16* __restrict__ outh,
                                 int D, int doElu, int total)
{
  int idx = blockIdx.x * blockDim.x + threadIdx.x;
  if (idx >= total) return;
  int d = idx % D;
  size_t bn = (size_t)(idx / D);
  float s = 0.f;
#pragma unroll
  for (int h = 0; h < 4; ++h) s += att[(bn * 4 + h) * D + d];
  float v = 0.25f * s + bias[d];
  if (doElu) v = v > 0.f ? v : (__expf(v) - 1.f);
  outf[idx] = v;
  if (outh) outh[idx] = (_Float16)v;
}

// gather X for GRU input GEMM: X[(t*B+b), n] = gat_out[b,n,t]
static __global__ void k_xgather(const float* __restrict__ go, _Float16* __restrict__ X16)
{
  int idx = blockIdx.x * blockDim.x + threadIdx.x;
  if (idx >= OFE * BB * NN) return;
  int nn  = idx % NN;
  int row = idx / NN;
  int t = row >> 6, b = row & 63;
  X16[idx] = (_Float16)go[((size_t)b * NN + nn) * OFE + t];
}

// ---------------------------------------------------------------------------
// GRU recurrence: single workgroup (32 waves), gh = h @ Whh^T via WMMA each step
// ---------------------------------------------------------------------------
static __global__ __launch_bounds__(1024)
void k_gru(const float* __restrict__ GI, const _Float16* __restrict__ whh16,
           const float* __restrict__ bhh, float* __restrict__ hbuf,
           float* __restrict__ ghbuf)
{
  __shared__ _Float16 h16s[64 * 264];
  const int tid = threadIdx.x, lane = tid & 31, wave = tid >> 5;
  for (int idx = tid; idx < 64 * HID; idx += 1024) hbuf[idx] = 0.f;
  __threadfence_block();
  __syncthreads();
  for (int t = 0; t < OFE; ++t) {
    for (int idx = tid; idx < 64 * HID; idx += 1024) {
      int b = idx >> 8, k = idx & 255;
      h16s[b * 264 + k] = (_Float16)hbuf[idx];
    }
    __syncthreads();
    // gh(64x768) = h(64x256) @ Whh^T : 192 tiles over 32 waves
#pragma unroll
    for (int q = 0; q < 6; ++q) {
      int id = wave * 6 + q;
      int mt = id / 48, nt = id % 48;
      v8f c = {};
      for (int kc = 0; kc < HID; kc += 32) {
        v16h af = frag_a(h16s + mt * 16 * 264 + kc, 264, lane);
        v16h bf = frag_b(whh16 + (size_t)(nt * 16) * HID + kc, HID, lane);
        c = wmma_f16(af, bf, c);
      }
      int row = mt * 16 + ((lane >> 4) << 3);
      int col = nt * 16 + (lane & 15);
#pragma unroll
      for (int v = 0; v < 8; ++v) ghbuf[(size_t)(row + v) * 768 + col] = c[v];
    }
    __threadfence_block();
    __syncthreads();
    const float* gi = GI + (size_t)t * 64 * 768;
    for (int idx = tid; idx < 64 * HID; idx += 1024) {
      int b = idx >> 8, k = idx & 255;
      float ir = gi[(size_t)b * 768 + k];
      float iz = gi[(size_t)b * 768 + 256 + k];
      float in_ = gi[(size_t)b * 768 + 512 + k];
      float hr = ghbuf[(size_t)b * 768 + k]       + bhh[k];
      float hz = ghbuf[(size_t)b * 768 + 256 + k] + bhh[256 + k];
      float hn = ghbuf[(size_t)b * 768 + 512 + k] + bhh[512 + k];
      float r  = 1.f / (1.f + __expf(-(ir + hr)));
      float zg = 1.f / (1.f + __expf(-(iz + hz)));
      float nn = tanhf(in_ + r * hn);
      hbuf[idx] = (1.f - zg) * nn + zg * hbuf[idx];
    }
    __threadfence_block();
    __syncthreads();
  }
}

// ---------------------------------------------------------------------------
// Per-node 2-layer MLP, fully fused, both GEMMs via WMMA. One block per node.
// ---------------------------------------------------------------------------
static __global__ __launch_bounds__(256)
void k_mlp(const float* __restrict__ gat_out, const float* __restrict__ hlast,
           const _Float16* __restrict__ w1t, const float* __restrict__ b1,
           const _Float16* __restrict__ w2t, const float* __restrict__ b2,
           float* __restrict__ pred)
{
  __shared__ _Float16 fuse[64 * 328];   // (B=64) x (320 pad 328)
  __shared__ _Float16 hm[64 * 168];     // (B=64) x (160 pad 168)
  const int n = blockIdx.x;
  const int tid = threadIdx.x, lane = tid & 31, wave = tid >> 5;
  for (int idx = tid; idx < 64 * MLP_IN; idx += 256) {
    int b = idx / MLP_IN, k = idx % MLP_IN;
    float v = (k < OFE) ? gat_out[((size_t)b * NN + n) * OFE + k]
                        : hlast[(size_t)b * HID + (k - OFE)];
    fuse[b * 328 + k] = (_Float16)v;
  }
  __syncthreads();
  const _Float16* w1n = w1t + (size_t)n * MLP_MID * MLP_IN;
#pragma unroll
  for (int q = 0; q < 5; ++q) {
    int id = wave * 5 + q;
    int mt = id / 10, nt = id % 10;
    v8f c = {};
    for (int kc = 0; kc < MLP_IN; kc += 32) {
      v16h af = frag_a(fuse + mt * 16 * 328 + kc, 328, lane);
      v16h bf = frag_b(w1n + (size_t)(nt * 16) * MLP_IN + kc, MLP_IN, lane);
      c = wmma_f16(af, bf, c);
    }
    int row = mt * 16 + ((lane >> 4) << 3);
    int col = nt * 16 + (lane & 15);
#pragma unroll
    for (int v = 0; v < 8; ++v) {
      float x = fmaxf(c[v] + b1[(size_t)n * MLP_MID + col], 0.f);
      hm[(row + v) * 168 + col] = (_Float16)x;
    }
  }
  __syncthreads();
  if (wave < 4) {
    const _Float16* w2n = w2t + (size_t)n * OUTD * MLP_MID;
    v8f c = {};
    for (int kc = 0; kc < MLP_MID; kc += 32) {
      v16h af = frag_a(hm + wave * 16 * 168 + kc, 168, lane);
      v16h bf = frag_b_guard(w2n + kc, MLP_MID, OUTD, lane);
      c = wmma_f16(af, bf, c);
    }
    int row = wave * 16 + ((lane >> 4) << 3);
    int o = lane & 15;
    if (o < OUTD) {
#pragma unroll
      for (int v = 0; v < 8; ++v)
        pred[((size_t)(row + v) * NN + n) * OUTD + o] = c[v] + b2[(size_t)n * OUTD + o];
    }
  }
}

// ---------------------------------------------------------------------------
// Workspace layout (bytes, all 256-aligned; big conv buffers recycled later)
// ---------------------------------------------------------------------------
static constexpr size_t alup(size_t x) { return (x + 255) & ~(size_t)255; }
static constexpr size_t SZ_Z1   = (size_t)NN * 8 * L1 * 4;       // 66,961,408
static constexpr size_t SZ_Z2   = (size_t)NN * 16 * L2 * 4;      // 133,627,904
static constexpr size_t SZ_Z2H  = (size_t)NN * 16 * L2 * 2;      // 66,813,952
static constexpr size_t SZ_FCWT = (size_t)EMB * DIMFC * 2;       // 13,049,600

static constexpr size_t OFF_Z1   = 0;
static constexpr size_t OFF_Z2   = OFF_Z1 + alup(SZ_Z1);
static constexpr size_t OFF_Z2H  = OFF_Z2 + alup(SZ_Z2);
static constexpr size_t OFF_FCWT = OFF_Z2H + alup(SZ_Z2H);
static constexpr size_t OFF_POOL = OFF_FCWT + alup(SZ_FCWT);

// small pool
static constexpr size_t O_ZRELU = OFF_POOL;                          // 512*100 f32
static constexpr size_t O_Z16   = O_ZRELU + alup(NN * EMB * 4);      // 512*128 f16
static constexpr size_t O_UTW   = O_Z16 + alup(NN * 128 * 2);
static constexpr size_t O_VTW   = O_UTW + alup(EMB * 128 * 2);
static constexpr size_t O_U     = O_VTW + alup(EMB * 128 * 2);
static constexpr size_t O_V     = O_U + alup(NN * EMB * 4);
static constexpr size_t O_ADJ   = O_V + alup(NN * EMB * 4);
static constexpr size_t CSZ     = alup((size_t)BB * NN * 4 * 4);     // coef arrays
static constexpr size_t O_ASRC1 = O_ADJ + alup((size_t)NN * NN * 4);
static constexpr size_t O_ADST1 = O_ASRC1 + CSZ;
static constexpr size_t O_M1    = O_ADST1 + CSZ;
static constexpr size_t O_S1    = O_M1 + CSZ;
static constexpr size_t O_ASRC2 = O_S1 + CSZ;
static constexpr size_t O_ADST2 = O_ASRC2 + CSZ;
static constexpr size_t O_M2    = O_ADST2 + CSZ;
static constexpr size_t O_S2    = O_M2 + CSZ;
static constexpr size_t O_G1WT  = O_S2 + CSZ;                        // 512x256 f16
static constexpr size_t O_G2WT  = O_G1WT + alup((size_t)NN * SEQ * 2);
static constexpr size_t O_GIH   = O_G2WT + alup((size_t)HID * HFE * 2);
static constexpr size_t O_WHH   = O_GIH + alup((size_t)768 * NN * 2);
static constexpr size_t O_HCUR  = O_WHH + alup((size_t)768 * HID * 2);
static constexpr size_t O_GH    = O_HCUR + alup((size_t)BB * HID * 4);
static constexpr size_t O_BN    = O_GH + alup((size_t)BB * 768 * 4); // 3x512 floats

// aliases into the big (dead-by-then) regions
static constexpr size_t O_ATT1  = OFF_Z1;                            // 67 MB
static constexpr size_t O_W1T   = OFF_Z1;                            // after att1 dead
static constexpr size_t O_W2T   = OFF_Z1 + alup((size_t)NN * MLP_MID * MLP_IN * 2);
static constexpr size_t O_X16   = OFF_Z2;                            // 16.8 MB
static constexpr size_t O_H116  = O_X16  + 16777216;                 // 33.5 MB
static constexpr size_t O_H216  = O_H116 + 33554432;                 // 16.8 MB
static constexpr size_t O_ATT2  = O_H216 + 16777216;                 // 33.5 MB
static constexpr size_t O_XG16  = O_ATT2 + 33554432;                 //  4.2 MB
static constexpr size_t O_GI    = O_XG16 + 4194304;                  // 12.6 MB
static constexpr size_t O_H1E   = OFF_Z2H;                           // 16.8 MB
static constexpr size_t O_H1E16 = O_H1E + 16777216;                  //  8.4 MB
static constexpr size_t O_GATO  = O_H1E16 + 8388608;                 //  8.4 MB

// ---------------------------------------------------------------------------
extern "C" void kernel_launch(void* const* d_in, const int* in_sizes, int n_in,
                              void* d_out, int out_size, void* d_ws, size_t ws_size,
                              hipStream_t stream) {
  (void)in_sizes; (void)n_in; (void)out_size; (void)ws_size;
  const float* data_x    = (const float*)d_in[0];
  const float* data_y    = (const float*)d_in[1];
  const float* node_feas = (const float*)d_in[2];
  const float* gumbel_u  = (const float*)d_in[3];
  const float* conv1_w = (const float*)d_in[5];
  const float* conv1_b = (const float*)d_in[6];
  const float* conv2_w = (const float*)d_in[7];
  const float* conv2_b = (const float*)d_in[8];
  const float* bn1_g = (const float*)d_in[9],  * bn1_b = (const float*)d_in[10];
  const float* bn2_g = (const float*)d_in[11], * bn2_b = (const float*)d_in[12];
  const float* bn3_g = (const float*)d_in[13], * bn3_b = (const float*)d_in[14];
  const float* fc_w = (const float*)d_in[15], * fc_b = (const float*)d_in[16];
  const float* fco_w = (const float*)d_in[17], * fco_b = (const float*)d_in[18];
  const float* fcc_w = (const float*)d_in[19], * fcc_b = (const float*)d_in[20];
  const float* gat1_w = (const float*)d_in[21];
  const float* gat1_as = (const float*)d_in[22], * gat1_ad = (const float*)d_in[23];
  const float* gat1_bs = (const float*)d_in[24];
  const float* gat2_w = (const float*)d_in[25];
  const float* gat2_as = (const float*)d_in[26], * gat2_ad = (const float*)d_in[27];
  const float* gat2_bs = (const float*)d_in[28];
  const float* gru_wih = (const float*)d_in[29], * gru_whh = (const float*)d_in[30];
  const float* gru_bih = (const float*)d_in[31], * gru_bhh = (const float*)d_in[32];
  const float* mlp_w1 = (const float*)d_in[33], * mlp_b1 = (const float*)d_in[34];
  const float* mlp_w2 = (const float*)d_in[35], * mlp_b2 = (const float*)d_in[36];

  char* ws = (char*)d_ws;
  auto F  = [&](size_t o) { return (float*)(ws + o); };
  auto H  = [&](size_t o) { return (_Float16*)(ws + o); };
  float* pred  = (float*)d_out;
  float* ytrue = (float*)d_out + BB * NN * OUTD;

  auto cdiv = [](int a, int b) { return (a + b - 1) / b; };

  // ---- stats regions zero (sums/sqs) -------------------------------------
  k_zero<<<6, 256, 0, stream>>>(F(O_BN), 3 * 512);

  // ---- conv1 -> bn1 ------------------------------------------------------
  k_conv1<<<cdiv(NN * L1, 256), 256, 0, stream>>>(node_feas, conv1_w, conv1_b, F(OFF_Z1));
  k_bn_stats<<<NN * 8, 256, 0, stream>>>(F(OFF_Z1), F(O_BN), F(O_BN) + 128, 8, L1);
  k_bn_finalize<<<1, 256, 0, stream>>>(F(O_BN), bn1_g, bn1_b, (float)NN * L1, 8);

  // ---- conv2 -> bn2 -> f16 GEMM operand ----------------------------------
  k_conv2<<<cdiv(NN * L2, 256), 256, 0, stream>>>(F(OFF_Z1), F(O_BN), conv2_w, conv2_b, F(OFF_Z2));
  k_bn_stats<<<NN * 16, 256, 0, stream>>>(F(OFF_Z2), F(O_BN) + 512, F(O_BN) + 512 + 128, 16, L2);
  k_bn_finalize<<<1, 256, 0, stream>>>(F(O_BN) + 512, bn2_g, bn2_b, (float)NN * L2, 16);
  k_z2h<<<cdiv(NN * 16 * L2, 256), 256, 0, stream>>>(F(OFF_Z2), F(O_BN) + 512, H(OFF_Z2H), NN * 16 * L2);

  // ---- fc GEMM (512 x 100 x 65248) + relu, then bn3 ----------------------
  k_transpose_b<<<cdiv(DIMFC * EMB, 256), 256, 0, stream>>>(fc_w, H(OFF_FCWT), 1, DIMFC, EMB);
  k_gemm<<<dim3(2, 8), 256, 0, stream>>>(H(OFF_Z2H), H(OFF_FCWT), F(O_ZRELU), nullptr,
                                         fc_b, NN, EMB, DIMFC, 1);
  k_bn3_stats<<<EMB, 256, 0, stream>>>(F(O_ZRELU), F(O_BN) + 1024, F(O_BN) + 1024 + 128);
  k_bn_finalize<<<1, 256, 0, stream>>>(F(O_BN) + 1024, bn3_g, bn3_b, (float)NN, EMB);
  k_z16pad<<<256, 256, 0, stream>>>(F(O_ZRELU), F(O_BN) + 1024, H(O_Z16));

  // ---- pair MLP as U/V GEMMs + fused gumbel adjacency --------------------
  k_uvw<<<cdiv(EMB * 128, 256), 256, 0, stream>>>(fco_w, H(O_UTW), H(O_VTW));
  k_gemm<<<dim3(2, 8), 256, 0, stream>>>(H(O_Z16), H(O_UTW), F(O_U), nullptr, nullptr, NN, EMB, 128, 0);
  k_gemm<<<dim3(2, 8), 256, 0, stream>>>(H(O_Z16), H(O_VTW), F(O_V), nullptr, nullptr, NN, EMB, 128, 0);
  k_pair<<<dim3(32, 32), 256, 0, stream>>>(F(O_U), F(O_V), fco_b, fcc_w, fcc_b, gumbel_u, F(O_ADJ));

  // ---- GAT layer 1 -------------------------------------------------------
  k_convert_h<<<cdiv(BB * NN * SEQ, 256), 256, 0, stream>>>(data_x, H(O_X16), BB * NN * SEQ);
  k_transpose_b<<<cdiv(SEQ * 512, 256), 256, 0, stream>>>(gat1_w, H(O_G1WT), 1, SEQ, 512);
  k_gemm<<<dim3(8, 512), 256, 0, stream>>>(H(O_X16), H(O_G1WT), nullptr, H(O_H116),
                                           nullptr, BB * NN, 512, SEQ, 0);
  k_attn_coefs<<<512, 256, 0, stream>>>(H(O_H116), gat1_as, gat1_ad, F(O_ASRC1), F(O_ADST1), HFE);
  k_gat_stats<<<dim3(NN, BB), 128, 0, stream>>>(F(O_ASRC1), F(O_ADST1), F(O_ADJ), F(O_M1), F(O_S1));
  k_gat_attn<HFE><<<dim3(8, 4, BB), 256, 0, stream>>>(H(O_H116), F(O_ASRC1), F(O_ADST1),
                                                      F(O_M1), F(O_S1), F(O_ADJ), F(O_ATT1));
  k_combine<<<cdiv(BB * NN * HFE, 256), 256, 0, stream>>>(F(O_ATT1), gat1_bs, F(O_H1E),
                                                          H(O_H1E16), HFE, 1, BB * NN * HFE);

  // ---- GAT layer 2 -------------------------------------------------------
  k_transpose_b<<<cdiv(HFE * HID, 256), 256, 0, stream>>>(gat2_w, H(O_G2WT), 1, HFE, HID);
  k_gemm<<<dim3(4, 512), 256, 0, stream>>>(H(O_H1E16), H(O_G2WT), nullptr, H(O_H216),
                                           nullptr, BB * NN, HID, HFE, 0);
  k_attn_coefs<<<512, 256, 0, stream>>>(H(O_H216), gat2_as, gat2_ad, F(O_ASRC2), F(O_ADST2), OFE);
  k_gat_stats<<<dim3(NN, BB), 128, 0, stream>>>(F(O_ASRC2), F(O_ADST2), F(O_ADJ), F(O_M2), F(O_S2));
  k_gat_attn<OFE><<<dim3(8, 4, BB), 256, 0, stream>>>(H(O_H216), F(O_ASRC2), F(O_ADST2),
                                                      F(O_M2), F(O_S2), F(O_ADJ), F(O_ATT2));
  k_combine<<<cdiv(BB * NN * OFE, 256), 256, 0, stream>>>(F(O_ATT2), gat2_bs, F(O_GATO),
                                                          nullptr, OFE, 0, BB * NN * OFE);

  // ---- GRU: hoisted input GEMM + in-workgroup WMMA recurrence ------------
  k_xgather<<<cdiv(OFE * BB * NN, 256), 256, 0, stream>>>(F(O_GATO), H(O_XG16));
  k_convert_h<<<cdiv(768 * NN, 256), 256, 0, stream>>>(gru_wih, H(O_GIH), 768 * NN);
  k_gemm<<<dim3(12, 64), 256, 0, stream>>>(H(O_XG16), H(O_GIH), F(O_GI), nullptr,
                                           gru_bih, OFE * BB, 768, NN, 0);
  k_convert_h<<<cdiv(768 * HID, 256), 256, 0, stream>>>(gru_whh, H(O_WHH), 768 * HID);
  k_gru<<<1, 1024, 0, stream>>>(F(O_GI), H(O_WHH), gru_bhh, F(O_HCUR), F(O_GH));

  // ---- per-node fused MLP (WMMA) + output --------------------------------
  k_transpose_b<<<cdiv(NN * MLP_IN * MLP_MID, 256), 256, 0, stream>>>(mlp_w1, H(O_W1T),
                                                                      NN, MLP_IN, MLP_MID);
  k_transpose_b<<<cdiv(NN * MLP_MID * OUTD, 256), 256, 0, stream>>>(mlp_w2, H(O_W2T),
                                                                    NN, MLP_MID, OUTD);
  k_mlp<<<NN, 256, 0, stream>>>(F(O_GATO), F(O_HCUR), H(O_W1T), mlp_b1, H(O_W2T), mlp_b2, pred);
  k_copy<<<cdiv(BB * NN * OUTD, 256), 256, 0, stream>>>(data_y, ytrue, BB * NN * OUTD);
}
